// WaveFormer_74844100100194
// MI455X (gfx1250) — matrix operand order repaired and checked by reading, not compiled, not run on hardware
//
#include <hip/hip_runtime.h>
#include <hip/hip_bf16.h>
#include <math.h>

// ---------------------------------------------------------------------------
// CDNA5 (gfx1250) WMMA types / helpers
// ---------------------------------------------------------------------------
typedef __bf16 bf16;
typedef __attribute__((ext_vector_type(16))) __bf16 bf16x16;
typedef __attribute__((ext_vector_type(8)))  float  v8f;
typedef __attribute__((ext_vector_type(4)))  int    int4v;

#define AS_GLOBAL __attribute__((address_space(1)))
#define AS_LDS    __attribute__((address_space(3)))

#if defined(__gfx1250__) && __has_builtin(__builtin_amdgcn_global_load_async_to_lds_b128)
#define HAVE_ASYNC_LDS 1
#endif

union FragU { bf16x16 v; uint4 q[2]; };

// A-matrix 16x32 bf16 fragment: lane L (m = L&15, g = L>>4)
// holds K = [k0+16g, k0+16g+16) contiguous in elems 0..15.
__device__ inline bf16x16 load_frag_a(const bf16* rowk /* &A[m][k0] */, int g) {
    FragU u;
    u.q[0] = *(const uint4*)(rowk + g * 16);
    u.q[1] = *(const uint4*)(rowk + g * 16 + 8);
    return u.v;
}
// B-matrix 32x16 bf16 fragment: lane L (n = L&15, g = L>>4) holds
// K = [k0+16g, k0+16g+16) contiguous from a K-contiguous [N][K] buffer.
__device__ inline bf16x16 load_frag_b(const bf16* colk /* &B[n][k0] */, int g) {
    FragU u;
    const bf16* p = colk + g * 16;
    u.q[0] = *(const uint4*)(p);
    u.q[1] = *(const uint4*)(p + 8);
    return u.v;
}

__device__ inline v8f wmma_bf16(bf16x16 a, bf16x16 b, v8f c) {
    // v_wmma_f32_16x16x32_bf16 (f32 accumulate)
    return __builtin_amdgcn_wmma_f32_16x16x32_bf16(
        /*neg_a=*/false, a, /*neg_b=*/false, b,
        /*c_mod=*/(short)0, c, /*reuse_a=*/false, /*reuse_b=*/false);
}

// ---------------------------------------------------------------------------
// Problem constants
// ---------------------------------------------------------------------------
#define BB 16
#define CC 384
#define HH 64
#define WW 64
#define MTOK (BB * HH * WW)                 // 65536 rows
#define NE   ((size_t)BB * HH * WW * CC)    // 25,165,824 elements
#define PLANE ((long)HH * WW * CC)          // 1,572,864 per batch
#define PI_F 3.14159265358979f

// ---------------------------------------------------------------------------
// Prep: DCT-II basis (bf16, row-major [freq][space], K-contiguous), its
// transpose, and log of the decay map.
// ---------------------------------------------------------------------------
__global__ void prep_kernel(bf16* Wn, bf16* WnT, float* logw) {
    for (int i = blockIdx.x * 256 + threadIdx.x; i < HH * HH; i += gridDim.x * 256) {
        int n = i >> 6, h = i & 63;
        float v = cosf((float)n * PI_F * ((float)h + 0.5f) / 64.f) * sqrtf(2.f / 64.f);
        if (n == 0) v *= 0.70710678118f;
        Wn[n * 64 + h]  = (bf16)v;
        WnT[h * 64 + n] = (bf16)v;
        float wn = PI_F * (float)n / 64.f;
        float wm = PI_F * (float)h / 64.f;
        logw[i] = -(wn * wn + wm * wm);     // log(exp(-(n^2+m^2)))
    }
}

__global__ void cast_bf16_kernel(const float* __restrict__ src, bf16* __restrict__ dst, int n) {
    for (int i = blockIdx.x * 256 + threadIdx.x; i < n; i += gridDim.x * 256)
        dst[i] = (bf16)src[i];
}

// ---------------------------------------------------------------------------
// Depthwise 3x3 conv (NCHW in, coalesced along w) -> NHWC bf16 (K-contiguous
// A matrix for the lin GEMM).
// ---------------------------------------------------------------------------
__global__ __launch_bounds__(256) void dwconv_kernel(
    const float* __restrict__ x, const float* __restrict__ w,
    const float* __restrict__ bias, bf16* __restrict__ xd) {
    size_t idx = (size_t)blockIdx.x * 256 + threadIdx.x;
    if (idx >= NE) return;
    int wp = (int)(idx & 63);
    int h  = (int)((idx >> 6) & 63);
    int cb = (int)(idx >> 12);
    int ch = cb % CC;
    int b  = cb / CC;
    const float* wf = w + ch * 9;
    float s = bias[ch];
    const size_t cbase = ((size_t)(b * CC + ch)) * (HH * WW);
    #pragma unroll
    for (int dy = 0; dy < 3; ++dy) {
        int yy = h + dy - 1;
        if (yy < 0 || yy >= HH) continue;
        #pragma unroll
        for (int dx = 0; dx < 3; ++dx) {
            int xx = wp + dx - 1;
            if (xx < 0 || xx >= WW) continue;
            s += x[cbase + (size_t)yy * WW + xx] * wf[dy * 3 + dx];
        }
    }
    xd[(((size_t)b * HH + h) * WW + wp) * CC + ch] = (bf16)s;
}

// ---------------------------------------------------------------------------
// Generic bf16 WMMA GEMM (K == 384):
//   out[m,n] = sum_k A[m*K+k] * Bw[n*K+k] + bias[n]
// Block tile 128x64 (8 waves, each 32x32 = 2x2 wmma tiles).
// The 64x384 bf16 weight panel is staged into LDS once per block with
// GLOBAL_LOAD_ASYNC_TO_LDS_B128 (ASYNCcnt) and reused across all 12 K steps.
// mode 0: lin -> n<384: bf16 xs ; n>=384: f32 silu(z)
// mode 1: tok -> f32 gelu
// mode 2: out -> f32 (+bias)
// ---------------------------------------------------------------------------
__global__ __launch_bounds__(256) void gemm_bf16_kernel(
    const bf16* __restrict__ A, const bf16* __restrict__ Bw,
    const float* __restrict__ bias, float* __restrict__ outF,
    bf16* __restrict__ outXs, float* __restrict__ outZ,
    int N, int K, int mode) {
    __shared__ bf16 bsh[64 * CC];                             // 48 KB weight panel

    // ---- async stage of the B panel: [64 rows of this block's n-range][K] ----
    const bf16* gsrc = Bw + (size_t)blockIdx.y * 64 * K;
    for (int i = threadIdx.x; i < (64 * CC) / 8; i += 256) {
#ifdef HAVE_ASYNC_LDS
        __builtin_amdgcn_global_load_async_to_lds_b128(
            (AS_GLOBAL int4v*)(gsrc + i * 8), (AS_LDS int4v*)(bsh + i * 8), 0, 0);
#else
        ((uint4*)bsh)[i] = ((const uint4*)gsrc)[i];
#endif
    }
#ifdef HAVE_ASYNC_LDS
#if __has_builtin(__builtin_amdgcn_s_wait_asynccnt)
    __builtin_amdgcn_s_wait_asynccnt(0);
#else
    asm volatile("s_wait_asynccnt 0" ::: "memory");
#endif
#endif
    __syncthreads();

    const int lane = threadIdx.x & 31;
    const int wv   = threadIdx.x >> 5;
    const int g    = lane >> 4, l16 = lane & 15;
    const int wm   = wv & 3, wn = wv >> 2;                    // 4 x 2 wave grid
    const int m_wave = blockIdx.x * 128 + wm * 32;
    const int n_loc  = wn * 32;                               // within block panel

    v8f acc[2][2] = {};
    const bf16* arow0 = A + (size_t)(m_wave + l16)      * K;
    const bf16* arow1 = A + (size_t)(m_wave + 16 + l16) * K;
    const bf16* bcol0 = bsh + (n_loc + l16)      * CC;
    const bf16* bcol1 = bsh + (n_loc + 16 + l16) * CC;

    for (int k = 0; k < K; k += 32) {
        __builtin_prefetch(arow0 + k + 64, 0, 1);             // global_prefetch
        bf16x16 a0 = load_frag_a(arow0 + k, g);
        bf16x16 a1 = load_frag_a(arow1 + k, g);
        bf16x16 b0 = load_frag_b(bcol0 + k, g);
        bf16x16 b1 = load_frag_b(bcol1 + k, g);
        acc[0][0] = wmma_bf16(a0, b0, acc[0][0]);
        acc[0][1] = wmma_bf16(a0, b1, acc[0][1]);
        acc[1][0] = wmma_bf16(a1, b0, acc[1][0]);
        acc[1][1] = wmma_bf16(a1, b1, acc[1][1]);
    }

    #pragma unroll
    for (int tm = 0; tm < 2; ++tm)
        #pragma unroll
        for (int tn = 0; tn < 2; ++tn) {
            const int col = blockIdx.y * 64 + n_loc + tn * 16 + l16;
            const float bv = bias[col];
            #pragma unroll
            for (int r = 0; r < 8; ++r) {
                const int m = m_wave + tm * 16 + r + 8 * g;
                float v = acc[tm][tn][r] + bv;
                if (mode == 0) {
                    if (col < CC) {
                        outXs[(size_t)m * CC + col] = (bf16)v;
                    } else {
                        outZ[(size_t)m * CC + (col - CC)] = v / (1.f + __expf(-v)); // silu
                    }
                } else if (mode == 1) {
                    outF[(size_t)m * CC + col] = 0.5f * v * (1.f + erff(v * 0.70710678f)); // gelu
                } else {
                    outF[(size_t)m * CC + col] = v;
                }
            }
        }
}

// ---------------------------------------------------------------------------
// 64-point (I)DCT along H or W:  dst[b,o,s,c] = sum_k T[o,k] * src[b,k(s),c]
// One block per (b, s): stages the 64x384 activation tile transposed into
// LDS ([c][k], K-contiguous) with b128 global loads + b16 LDS scatter.
// Basis A fragments hoisted per o-tile (reused over 3 column tiles / wave).
// mode 1 fuses the cos/sin/decay modulation (x_u0 == x_v0 in the reference,
// so it collapses to a single per-element scale); mode 2 stores f32.
// ---------------------------------------------------------------------------
#define LDS_STRIDE 72
__global__ __launch_bounds__(256) void dct64_kernel(
    const bf16* __restrict__ src, void* __restrict__ dstv, const bf16* __restrict__ T,
    long srcStrideS, long srcStrideK, long dstStrideS, long dstStrideO,
    int mode, const float* __restrict__ tmod, const float* __restrict__ logw,
    const float* __restrict__ cptr, const float* __restrict__ aptr) {
    __shared__ bf16 lds[CC * LDS_STRIDE];
    const int s = blockIdx.x, b = blockIdx.y;
    const long base  = (long)b * PLANE + (long)s * srcStrideS;
    const long dbase = (long)b * PLANE + (long)s * dstStrideS;

    // Stage + transpose: 16B vector loads along c, b16 scatter into [c][k].
    for (int i = threadIdx.x; i < (64 * CC) / 8; i += 256) {
        const int c8 = i % (CC / 8);
        const int k  = i / (CC / 8);
        FragU u;
        u.q[0] = *(const uint4*)(src + base + (long)k * srcStrideK + c8 * 8);
        const bf16* e = (const bf16*)&u.q[0];
        const int cb = c8 * 8;
        #pragma unroll
        for (int j = 0; j < 8; ++j)
            lds[(cb + j) * LDS_STRIDE + k] = e[j];
    }
    __syncthreads();

    float ccv = 0.f, halfalpha = 0.f;
    if (mode == 1) { ccv = cptr[0]; halfalpha = 0.5f * aptr[0]; }

    const int lane = threadIdx.x & 31, wv = threadIdx.x >> 5;
    const int g = lane >> 4, l16 = lane & 15;

    #pragma unroll
    for (int ot = 0; ot < 4; ++ot) {                          // 4 o-tiles
        const int o0 = ot * 16;
        const bf16* arow = T + (o0 + l16) * 64;
        const bf16x16 a0 = load_frag_a(arow, g);              // K 0..31
        const bf16x16 a1 = load_frag_a(arow + 32, g);         // K 32..63
        for (int ct = wv; ct < 24; ct += 8) {                 // 3 col tiles / wave
            const int c0 = ct * 16;
            const bf16* bcol = lds + (c0 + l16) * LDS_STRIDE;
            v8f acc = {};
            acc = wmma_bf16(a0, load_frag_b(bcol, g), acc);
            acc = wmma_bf16(a1, load_frag_b(bcol + 32, g), acc);

            const int col = c0 + l16;
            #pragma unroll
            for (int r = 0; r < 8; ++r) {
                const int m = o0 + r + 8 * g;
                const long di = dbase + (long)m * dstStrideO + col;
                float v = acc[r];
                if (mode == 1) {
                    float tv = tmod[di];
                    float ct2 = ccv * tv;
                    v *= (cosf(ct2) + sinf(ct2) * (1.f + halfalpha) / (ccv + 1e-8f))
                         * __expf(tv * logw[s * 64 + m]);
                    ((bf16*)dstv)[di] = (bf16)v;
                } else if (mode == 2) {
                    ((float*)dstv)[di] = v;
                } else {
                    ((bf16*)dstv)[di] = (bf16)v;
                }
            }
        }
    }
}

// ---------------------------------------------------------------------------
// Fused LayerNorm(C) * silu(z) -> bf16 rows for the final GEMM.
// One 128-thread block per token row (3 elements/thread).
// ---------------------------------------------------------------------------
__global__ __launch_bounds__(128) void ln_silu_kernel(
    const float* __restrict__ xo, const float* __restrict__ zs,
    const float* __restrict__ gam, const float* __restrict__ bet,
    bf16* __restrict__ gy) {
    __shared__ float red[128];
    const size_t row = blockIdx.x;
    const float* xr = xo + row * CC;
    const int t = threadIdx.x;
    float x0 = xr[t], x1 = xr[t + 128], x2 = xr[t + 256];

    red[t] = x0 + x1 + x2;
    __syncthreads();
    for (int off = 64; off; off >>= 1) {
        if (t < off) red[t] += red[t + off];
        __syncthreads();
    }
    const float mu = red[0] * (1.f / CC);
    __syncthreads();

    float d0 = x0 - mu, d1 = x1 - mu, d2 = x2 - mu;
    red[t] = d0 * d0 + d1 * d1 + d2 * d2;
    __syncthreads();
    for (int off = 64; off; off >>= 1) {
        if (t < off) red[t] += red[t + off];
        __syncthreads();
    }
    const float rs = rsqrtf(red[0] * (1.f / CC) + 1e-5f);

    const float* zr = zs + row * CC;
    #pragma unroll
    for (int i = 0; i < 3; ++i) {
        int c = t + i * 128;
        float xv = (i == 0) ? x0 : (i == 1) ? x1 : x2;
        float y = (xv - mu) * rs * gam[c] + bet[c];
        gy[row * CC + c] = (bf16)(y * zr[c]);
    }
}

// ---------------------------------------------------------------------------
// Launch
// ---------------------------------------------------------------------------
extern "C" void kernel_launch(void* const* d_in, const int* in_sizes, int n_in,
                              void* d_out, int out_size, void* d_ws, size_t ws_size,
                              hipStream_t stream) {
    const float* x_in   = (const float*)d_in[0];
    const float* fe_in  = (const float*)d_in[1];
    const float* dw_w   = (const float*)d_in[2];
    const float* dw_b   = (const float*)d_in[3];
    const float* lin_w  = (const float*)d_in[4];
    const float* lin_b  = (const float*)d_in[5];
    const float* tok_w  = (const float*)d_in[6];
    const float* tok_b  = (const float*)d_in[7];
    const float* ln_g   = (const float*)d_in[8];
    const float* ln_bb  = (const float*)d_in[9];
    const float* out_w  = (const float*)d_in[10];
    const float* out_b  = (const float*)d_in[11];
    const float* c_s    = (const float*)d_in[12];
    const float* alph_s = (const float*)d_in[13];

    // Workspace carve-up
    bf16*  xd   = (bf16*)d_ws;           // dwconv out, NHWC bf16
    bf16*  fe   = xd   + NE;             // freq_embed bf16
    bf16*  xs   = fe   + NE;             // lin out (first C) bf16
    bf16*  bufA = xs   + NE;             // DCT ping
    bf16*  bufB = bufA + NE;             // DCT pong
    bf16*  gy   = bufB + NE;             // LN*silu rows bf16
    float* zsil = (float*)(gy + NE);     // silu(z) f32
    float* tmod = zsil + NE;             // gelu(tok) f32
    float* xo   = tmod + NE;             // IDCT result f32
    bf16*  linw = (bf16*)(xo + NE);
    bf16*  tokw = linw + 2 * CC * CC;
    bf16*  outw = tokw + CC * CC;
    bf16*  Wn   = outw + CC * CC;        // cos basis [n][h]
    bf16*  WnT  = Wn   + 64 * 64;        // transposed basis [h][n]
    float* logw = (float*)(WnT + 64 * 64);

    prep_kernel<<<16, 256, 0, stream>>>(Wn, WnT, logw);
    cast_bf16_kernel<<<1152, 256, 0, stream>>>(lin_w, linw, 2 * CC * CC);
    cast_bf16_kernel<<<576,  256, 0, stream>>>(tok_w, tokw, CC * CC);
    cast_bf16_kernel<<<576,  256, 0, stream>>>(out_w, outw, CC * CC);
    cast_bf16_kernel<<<4096, 256, 0, stream>>>(fe_in, fe, (int)NE);

    dwconv_kernel<<<(int)(NE / 256), 256, 0, stream>>>(x_in, dw_w, dw_b, xd);

    // lin: [65536 x 768] = xd @ lin_w^T  -> xs (bf16), silu(z) (f32)
    gemm_bf16_kernel<<<dim3(MTOK / 128, 768 / 64), 256, 0, stream>>>(
        xd, linw, lin_b, nullptr, xs, zsil, 768, CC, 0);
    // tok: t = gelu(fe @ tok_w^T)
    gemm_bf16_kernel<<<dim3(MTOK / 128, CC / 64), 256, 0, stream>>>(
        fe, tokw, tok_b, tmod, nullptr, nullptr, CC, CC, 1);

    const dim3 dgrid(64, BB);
    // DCT over H: contract dim1 (stride 24576), spectator w (stride 384)
    dct64_kernel<<<dgrid, 256, 0, stream>>>(xs, bufA, Wn,
        384, 24576, 384, 24576, 0, nullptr, nullptr, nullptr, nullptr);
    // DCT over W + fused cos/sin/decay modulation
    dct64_kernel<<<dgrid, 256, 0, stream>>>(bufA, bufB, Wn,
        24576, 384, 24576, 384, 1, tmod, logw, c_s, alph_s);
    // IDCT over H (basis transposed)
    dct64_kernel<<<dgrid, 256, 0, stream>>>(bufB, bufA, WnT,
        384, 24576, 384, 24576, 0, nullptr, nullptr, nullptr, nullptr);
    // IDCT over W -> f32 spatial result
    dct64_kernel<<<dgrid, 256, 0, stream>>>(bufA, xo, WnT,
        24576, 384, 24576, 384, 2, nullptr, nullptr, nullptr, nullptr);

    ln_silu_kernel<<<MTOK, 128, 0, stream>>>(xo, zsil, ln_g, ln_bb, gy);

    // out: d_out = gy @ out_w^T + out_b  (f32)
    gemm_bf16_kernel<<<dim3(MTOK / 128, CC / 64), 256, 0, stream>>>(
        gy, outw, out_b, (float*)d_out, nullptr, nullptr, CC, CC, 2);
}
